// DND_50010599195076
// MI455X (gfx1250) — matrix operand order repaired and checked by reading, not compiled, not run on hardware
//
#include <hip/hip_runtime.h>
#include <hip/hip_bf16.h>
#include <stdint.h>

#define DDIM     64
#define KSEL     50
#define NSLICE   32
#define CAP      128
#define NBINS    1024
#define NWAVES   4

typedef __attribute__((ext_vector_type(16))) __bf16 v16bf;
typedef __attribute__((ext_vector_type(8)))  float  v8f;

union Frag { v16bf v; uint4 u[2]; };

// round-to-nearest-even f32 -> bf16, packed pair
__device__ __forceinline__ unsigned pack2bf(float lo, float hi) {
    unsigned a = __float_as_uint(lo), b = __float_as_uint(hi);
    a += 0x7FFFu + ((a >> 16) & 1u);
    b += 0x7FFFu + ((b >> 16) & 1u);
    return (a >> 16) | (b & 0xFFFF0000u);
}

// branchless order-preserving float -> uint (ascending): 3 VALU ops
__device__ __forceinline__ unsigned orderKey(float f) {
    unsigned u = __float_as_uint(f);
    return u ^ (unsigned)(((int)u >> 31) | (int)0x80000000);
}

// exact inverse of orderKey
__device__ __forceinline__ float unorderKey(unsigned k) {
    unsigned u = (k & 0x80000000u) ? (k ^ 0x80000000u) : ~k;
    return __uint_as_float(u);
}

// ---------------------------------------------------------------------------
// Kernel 0: table_keys f32 -> bf16 (row-major, 128B/row) + row squared norms.
// One-time pass; thereafter the 64MB bf16 table lives in the 192MB L2.
// ---------------------------------------------------------------------------
__global__ __launch_bounds__(256) void DND_prep_kernel(
    const float* __restrict__ tk, uint4* __restrict__ tbf,
    float* __restrict__ norms, int C) {
    int c = blockIdx.x * 256 + threadIdx.x;
    if (c >= C) return;
    const float4* row = (const float4*)(tk + (size_t)c * DDIM);
    float nrm = 0.0f;
    uint4* dst = tbf + (size_t)c * 8;
#pragma unroll
    for (int i = 0; i < 8; ++i) {
        float4 a = row[2 * i], b = row[2 * i + 1];
        nrm += a.x * a.x + a.y * a.y + a.z * a.z + a.w * a.w;
        nrm += b.x * b.x + b.y * b.y + b.z * b.z + b.w * b.w;
        dst[i] = make_uint4(pack2bf(a.x, a.y), pack2bf(a.z, a.w),
                            pack2bf(b.x, b.y), pack2bf(b.z, b.w));
    }
    norms[c] = nrm;
}

// ---------------------------------------------------------------------------
// Kernel 1: fused WMMA distances + histogram top-64 select per (tile, slice).
// Block = 128 threads (4 wave32s), 16 queries. 2 column-chunks (32 table rows,
// 4 WMMAs) per loop iteration. score = ||c||^2 - 2<q,c> (query norm dropped).
// Pass 1: branch-free 1024-bin histogram (invalid lanes -> +INF -> bin 1022).
// Pass 2: single float-threshold compare per element, rare LDS append.
// ---------------------------------------------------------------------------
__global__ __launch_bounds__(128, 1) void DND_knn_slice_kernel(
    const float* __restrict__ keys, const uint4* __restrict__ tbf,
    const float* __restrict__ norms,
    float* __restrict__ candDist, unsigned* __restrict__ candIdx,
    unsigned* __restrict__ candCnt, int C, int sliceLen) {

    __shared__ __align__(16) unsigned hist[16][NBINS / 2];
    __shared__ float    cd[16][CAP];
    __shared__ unsigned ci[16][CAP];
    __shared__ unsigned cnt[16];
    __shared__ float    thrF[16];

    const int tid   = threadIdx.x;
    const int lane  = tid & 31;
    const int wave  = tid >> 5;
    const int n     = lane & 15;   // table row within chunk / output column
    const int kSel  = lane >> 4;   // which K-half this lane carries
    const int qBase = blockIdx.x * 16;
    const int s0    = blockIdx.y * sliceLen;
    const int s1    = min(s0 + sliceLen, C);
    const int nChunks = (s1 - s0 + 15) >> 4;
    const float INFV = __builtin_inff();

    // A fragments (16x32 bf16 each) for K 0..31 / 32..63, contiguous k = 16*kSel + s order (B matches).
    Frag a0, a1;
    {
        const float* qrow = keys + (size_t)(qBase + n) * DDIM;
#pragma unroll
        for (int h = 0; h < 2; ++h) {
            const int kb = h * 32;
            const float4* g0 = (const float4*)(qrow + kb + kSel * 16);
            const float4* g1 = (const float4*)(qrow + kb + kSel * 16 + 8);
            float4 p0 = g0[0], p1 = g0[1], p2 = g1[0], p3 = g1[1];
            uint4 lo = make_uint4(pack2bf(p0.x, p0.y), pack2bf(p0.z, p0.w),
                                  pack2bf(p1.x, p1.y), pack2bf(p1.z, p1.w));
            uint4 hi = make_uint4(pack2bf(p2.x, p2.y), pack2bf(p2.z, p2.w),
                                  pack2bf(p3.x, p3.y), pack2bf(p3.z, p3.w));
            if (h == 0) { a0.u[0] = lo; a0.u[1] = hi; }
            else        { a1.u[0] = lo; a1.u[1] = hi; }
        }
    }

    {   // vectorized histogram clear: 16*512 words = 2048 uint4
        uint4 z = make_uint4(0u, 0u, 0u, 0u);
        for (int i = tid; i < 16 * (NBINS / 2) / 4; i += 128) ((uint4*)hist)[i] = z;
    }
    __syncthreads();

    unsigned* hbase = &hist[kSel * 8][0];

    // ---------------- PASS 1: branch-free histogram ----------------
    for (int base = wave * 2; base < nChunks; base += 2 * NWAVES) {
        int  c0 = s0 + base * 16 + n;
        int  c1 = c0 + 16;
        bool v0 = c0 < s1, v1 = c1 < s1;
        int  cc0 = v0 ? c0 : (C - 1);
        int  cc1 = v1 ? c1 : (C - 1);
        const uint4* rp0 = tbf + (size_t)cc0 * 8;
        const uint4* rp1 = tbf + (size_t)cc1 * 8;
        Frag b0a, b0b, b1a, b1b;   // lanes 0-15: K lo half, lanes 16-31: K hi
        b0a.u[0] = rp0[kSel * 2];     b0a.u[1] = rp0[kSel * 2 + 1];
        b0b.u[0] = rp0[4 + kSel * 2]; b0b.u[1] = rp0[5 + kSel * 2];
        b1a.u[0] = rp1[kSel * 2];     b1a.u[1] = rp1[kSel * 2 + 1];
        b1b.u[0] = rp1[4 + kSel * 2]; b1b.u[1] = rp1[5 + kSel * 2];
        float nrm0 = norms[cc0];
        float nrm1 = norms[cc1];
        // unconditional prefetch, address clamped into the slice
        int cn0 = min(c0 + 32 * NWAVES, s1 - 1);
        int cn1 = min(c1 + 32 * NWAVES, s1 - 1);
        __builtin_prefetch((const void*)(tbf + (size_t)cn0 * 8), 0, 1);
        __builtin_prefetch((const void*)(tbf + (size_t)cn1 * 8), 0, 1);

        v8f acc0 = {}, acc1 = {};
        acc0 = __builtin_amdgcn_wmma_f32_16x16x32_bf16(false, a0.v, false, b0a.v,
                                                       (short)0, acc0, false, false);
        acc0 = __builtin_amdgcn_wmma_f32_16x16x32_bf16(false, a1.v, false, b0b.v,
                                                       (short)0, acc0, false, false);
        acc1 = __builtin_amdgcn_wmma_f32_16x16x32_bf16(false, a0.v, false, b1a.v,
                                                       (short)0, acc1, false, false);
        acc1 = __builtin_amdgcn_wmma_f32_16x16x32_bf16(false, a1.v, false, b1b.v,
                                                       (short)0, acc1, false, false);
#pragma unroll
        for (int r = 0; r < 8; ++r) {
            // invalid lanes score +INF -> land in bin 1022, never reach the
            // threshold scan (cum >= 64 is hit far below); keeps this block
            // straight-line (no exec-mask churn).
            float sA = v0 ? __builtin_fmaf(-2.0f, acc0[r], nrm0) : INFV;
            unsigned binA = orderKey(sA) >> 22;
            atomicAdd(hbase + r * (NBINS / 2) + (binA >> 1),
                      1u << ((binA & 1u) << 4));
            float sB = v1 ? __builtin_fmaf(-2.0f, acc1[r], nrm1) : INFV;
            unsigned binB = orderKey(sB) >> 22;
            atomicAdd(hbase + r * (NBINS / 2) + (binB >> 1),
                      1u << ((binB & 1u) << 4));
        }
    }
    __syncthreads();

    if (tid < 16) {   // first bin where cumulative count >= 64 -> float edge
        unsigned cum = 0, t = NBINS - 1;
        for (int b2 = 0; b2 < NBINS / 2; ++b2) {
            unsigned h = hist[tid][b2];
            cum += h & 0xFFFFu;
            if (cum >= 64u) { t = 2u * b2; break; }
            cum += h >> 16;
            if (cum >= 64u) { t = 2u * b2 + 1u; break; }
        }
        thrF[tid] = (t >= NBINS - 1) ? __builtin_inff()
                                     : unorderKey((t + 1u) << 22);
        cnt[tid] = 0u;
    }
    __syncthreads();

    float thrL[8];
#pragma unroll
    for (int r = 0; r < 8; ++r) thrL[r] = thrF[kSel * 8 + r];

    // ---------------- PASS 2: collect (1 fma + 1 cmp per element) ----------
    for (int base = wave * 2; base < nChunks; base += 2 * NWAVES) {
        int  c0 = s0 + base * 16 + n;
        int  c1 = c0 + 16;
        bool v0 = c0 < s1, v1 = c1 < s1;
        int  cc0 = v0 ? c0 : (C - 1);
        int  cc1 = v1 ? c1 : (C - 1);
        const uint4* rp0 = tbf + (size_t)cc0 * 8;
        const uint4* rp1 = tbf + (size_t)cc1 * 8;
        Frag b0a, b0b, b1a, b1b;
        b0a.u[0] = rp0[kSel * 2];     b0a.u[1] = rp0[kSel * 2 + 1];
        b0b.u[0] = rp0[4 + kSel * 2]; b0b.u[1] = rp0[5 + kSel * 2];
        b1a.u[0] = rp1[kSel * 2];     b1a.u[1] = rp1[kSel * 2 + 1];
        b1b.u[0] = rp1[4 + kSel * 2]; b1b.u[1] = rp1[5 + kSel * 2];
        float nrm0 = norms[cc0];
        float nrm1 = norms[cc1];

        v8f acc0 = {}, acc1 = {};
        acc0 = __builtin_amdgcn_wmma_f32_16x16x32_bf16(false, a0.v, false, b0a.v,
                                                       (short)0, acc0, false, false);
        acc0 = __builtin_amdgcn_wmma_f32_16x16x32_bf16(false, a1.v, false, b0b.v,
                                                       (short)0, acc0, false, false);
        acc1 = __builtin_amdgcn_wmma_f32_16x16x32_bf16(false, a0.v, false, b1a.v,
                                                       (short)0, acc1, false, false);
        acc1 = __builtin_amdgcn_wmma_f32_16x16x32_bf16(false, a1.v, false, b1b.v,
                                                       (short)0, acc1, false, false);
#pragma unroll
        for (int r = 0; r < 8; ++r) {
            int m = kSel * 8 + r;
            // +INF redirect makes the insert condition a single float compare
            float sA = v0 ? __builtin_fmaf(-2.0f, acc0[r], nrm0) : INFV;
            if (sA < thrL[r]) {
                unsigned pos = atomicAdd(&cnt[m], 1u);
                if (pos < CAP) { cd[m][pos] = sA; ci[m][pos] = (unsigned)c0; }
            }
            float sB = v1 ? __builtin_fmaf(-2.0f, acc1[r], nrm1) : INFV;
            if (sB < thrL[r]) {
                unsigned pos = atomicAdd(&cnt[m], 1u);
                if (pos < CAP) { cd[m][pos] = sB; ci[m][pos] = (unsigned)c1; }
            }
        }
    }
    __syncthreads();

    const int m  = tid >> 3;   // 8 threads per query
    const int t8 = tid & 7;
    unsigned cm = min(cnt[m], (unsigned)CAP);
    size_t gbase = ((size_t)(qBase + m) * NSLICE + blockIdx.y) * CAP;
    for (unsigned k = (unsigned)t8; k < cm; k += 8u) {
        candDist[gbase + k] = cd[m][k];
        candIdx[gbase + k]  = ci[m][k];
    }
    if (t8 == 0) candCnt[(size_t)(qBase + m) * NSLICE + blockIdx.y] = cm;
}

// ---------------------------------------------------------------------------
// Kernel 2: per-query merge -> exact top-50 -> exact f32 re-scored weights.
// Argmin rounds use wave32 shfl_xor reductions (2 barriers/round).
// ---------------------------------------------------------------------------
__global__ __launch_bounds__(256) void DND_knn_merge_kernel(
    const float* __restrict__ keys, const float* __restrict__ tk,
    const float* __restrict__ tv, const float* __restrict__ candDist,
    const unsigned* __restrict__ candIdx, const unsigned* __restrict__ candCnt,
    float* __restrict__ out) {

    __shared__ float    sd[NSLICE * CAP];
    __shared__ unsigned si[NSLICE * CAP];
    __shared__ int      offs[NSLICE + 1];
    __shared__ float    rv[8];
    __shared__ int      ri[8];
    __shared__ unsigned selIdx[KSEL];
    __shared__ float    qv[DDIM];
    __shared__ float    part[KSEL * 4];
    __shared__ float    wgt[KSEL];

    const int q = blockIdx.x, tid = threadIdx.x;
    const int lane = tid & 31, wid = tid >> 5;
    const float INFV = __builtin_inff();

    if (tid == 0) {
        int off = 0;
        for (int s = 0; s < NSLICE; ++s) {
            offs[s] = off;
            off += (int)candCnt[(size_t)q * NSLICE + s];
        }
        offs[NSLICE] = off;
    }
    if (tid < DDIM) qv[tid] = keys[(size_t)q * DDIM + tid];
    __syncthreads();

    const int nTot = offs[NSLICE];
    for (int s = 0; s < NSLICE; ++s) {
        int b = offs[s], cm = offs[s + 1] - b;
        size_t g = ((size_t)q * NSLICE + s) * CAP;
        for (int k = tid; k < cm; k += 256) {
            sd[b + k] = candDist[g + k];
            si[b + k] = candIdx[g + k];
        }
    }
    __syncthreads();

    const int rounds = min(KSEL, nTot);
    for (int rsel = 0; rsel < rounds; ++rsel) {
        float best = INFV; int bp = 0;
        for (int k = tid; k < nTot; k += 256)
            if (sd[k] < best) { best = sd[k]; bp = k; }
#pragma unroll
        for (int off = 16; off > 0; off >>= 1) {   // wave32 argmin
            float ov = __shfl_xor(best, off, 32);
            int   oi = __shfl_xor(bp,   off, 32);
            if (ov < best) { best = ov; bp = oi; }
        }
        if (lane == 0) { rv[wid] = best; ri[wid] = bp; }
        __syncthreads();
        if (tid == 0) {
            float b = rv[0]; int p = ri[0];
            for (int w = 1; w < 8; ++w)
                if (rv[w] < b) { b = rv[w]; p = ri[w]; }
            selIdx[rsel] = si[p];
            sd[p] = INFV;
        }
        __syncthreads();
    }

    if (tid < rounds * 4) {      // exact f32 squared distances, 4 threads/neigh
        int j = tid >> 2, seg = tid & 3;
        const float* nb = tk + (size_t)selIdx[j] * DDIM;
        float s = 0.0f;
#pragma unroll
        for (int d = 0; d < 16; ++d) {
            float df = qv[seg * 16 + d] - nb[seg * 16 + d];
            s += df * df;
        }
        part[tid] = s;
    }
    __syncthreads();
    if (tid < rounds) {
        float sq = part[tid * 4] + part[tid * 4 + 1] +
                   part[tid * 4 + 2] + part[tid * 4 + 3];
        wgt[tid] = 1.0f / (sq + 1e-3f);
    }
    __syncthreads();
    if (tid == 0) {
        float sw = 0.0f, sv = 0.0f;
        for (int j = 0; j < rounds; ++j) {
            sw += wgt[j];
            sv += wgt[j] * tv[selIdx[j]];
        }
        out[q] = sv / sw;
    }
}

// ---------------------------------------------------------------------------
extern "C" void kernel_launch(void* const* d_in, const int* in_sizes, int n_in,
                              void* d_out, int out_size, void* d_ws, size_t ws_size,
                              hipStream_t stream) {
    const float* keys = (const float*)d_in[0];
    const float* tk   = (const float*)d_in[1];
    const float* tv   = (const float*)d_in[2];
    float*       out  = (float*)d_out;

    const int B = in_sizes[0] / DDIM;   // 1024
    const int C = in_sizes[2];          // 500000

    // Workspace: bf16 table (C*128B) | norms | candDist | candIdx | candCnt
    char*  ws = (char*)d_ws;
    size_t o  = 0;
    uint4* tbf = (uint4*)(ws + o);      o += (size_t)C * DDIM * 2;  o = (o + 255) & ~(size_t)255;
    float* norms = (float*)(ws + o);    o += (size_t)C * 4;         o = (o + 255) & ~(size_t)255;
    float* candDist = (float*)(ws + o); o += (size_t)B * NSLICE * CAP * 4; o = (o + 255) & ~(size_t)255;
    unsigned* candIdx = (unsigned*)(ws + o); o += (size_t)B * NSLICE * CAP * 4; o = (o + 255) & ~(size_t)255;
    unsigned* candCnt = (unsigned*)(ws + o);

    const int sliceLen = (C + NSLICE - 1) / NSLICE;

    DND_prep_kernel<<<(C + 255) / 256, 256, 0, stream>>>(tk, tbf, norms, C);

    DND_knn_slice_kernel<<<dim3(B / 16, NSLICE), 128, 0, stream>>>(
        keys, tbf, norms, candDist, candIdx, candCnt, C, sliceLen);

    DND_knn_merge_kernel<<<B, 256, 0, stream>>>(
        keys, tk, tv, candDist, candIdx, candCnt, out);
}
